// Transformer_Block_40441412059263
// MI455X (gfx1250) — matrix operand order repaired and checked by reading, not compiled, not run on hardware
//
#include <hip/hip_runtime.h>
#include <hip/hip_bf16.h>

typedef __attribute__((ext_vector_type(16))) __bf16 bf16x16;
typedef __attribute__((ext_vector_type(8)))  __bf16 bf16x8;
typedef __attribute__((ext_vector_type(4)))  __bf16 bf16x4;
typedef __attribute__((ext_vector_type(8)))  float  f32x8;
typedef __attribute__((ext_vector_type(4)))  float  f32x4;

#define LDSK 40  // padded LDS row stride (elements) for 32-wide K tile; 80B rows stay 16B aligned

__device__ __forceinline__ __bf16 to_bf16(float f) {
    unsigned u = __builtin_bit_cast(unsigned, f);
    unsigned r = u + 0x7FFFu + ((u >> 16) & 1u);   // round-to-nearest-even
    unsigned short h = (unsigned short)(r >> 16);
    return __builtin_bit_cast(__bf16, h);
}

// ---------------------------------------------------------------------------
// Uniform NT GEMM: C = act( scale * A[M,K] x Bt[N,K]^T + bias )
// A, Bt are bf16 row-major; accumulate f32 via v_wmma_f32_16x16x32_bf16.
// Block tile 128x128, 256 threads (8 waves), wave tile 32x64 (8 WMMA tiles).
// Outputs (template-selected): f32 C, bf16 C, and/or bf16 C^T (per-batch
// transposed: Cbt[(gm/trows)*N + gn][gm%trows]).
// ---------------------------------------------------------------------------
template<int CAUSAL, int BIASRELU, int OUTF32, int OUTB16, int OUTB16T>
__global__ __launch_bounds__(256)
void gemm_bf16_nt(const __bf16* __restrict__ Ag, const __bf16* __restrict__ Btg,
                  const float* __restrict__ bias,
                  float* __restrict__ Cf, __bf16* __restrict__ Cb, __bf16* __restrict__ Cbt,
                  int M, int N, int K, int lda, int ldb, int ldc,
                  long sA, long sB, long sC, float scale, int trows)
{
    __shared__ __bf16 As[128 * LDSK];
    __shared__ __bf16 Bs[128 * LDSK];

    const int bz = blockIdx.z;
    const __bf16* A  = Ag  + (long)bz * sA;
    const __bf16* Bt = Btg + (long)bz * sB;

    const int m0 = blockIdx.y * 128;
    const int n0 = blockIdx.x * 128;

    const int tid  = threadIdx.x;
    const int wave = tid >> 5;
    const int lane = tid & 31;
    const int l16  = lane & 15;
    const int hi   = lane >> 4;
    const int wm   = (wave >> 1) * 32;   // wave row offset: 0,32,64,96
    const int wn   = (wave & 1)  * 64;   // wave col offset: 0,64

    // staging assignment: 512 16B chunks per operand tile, 2 per thread
    const int srow0 = (tid + 0)   >> 2, sc0 = ((tid + 0)   & 3) << 3;
    const int srow1 = (tid + 256) >> 2, sc1 = ((tid + 256) & 3) << 3;

    f32x8 acc[2][4] = {};

    for (int k0 = 0; k0 < K; k0 += 32) {
        // ---- stage A/Bt tiles (pure 16B copies, no conversion) ----
        *(bf16x8*)&As[srow0 * LDSK + sc0] = *(const bf16x8*)(A  + (long)(m0 + srow0) * lda + k0 + sc0);
        *(bf16x8*)&As[srow1 * LDSK + sc1] = *(const bf16x8*)(A  + (long)(m0 + srow1) * lda + k0 + sc1);
        *(bf16x8*)&Bs[srow0 * LDSK + sc0] = *(const bf16x8*)(Bt + (long)(n0 + srow0) * ldb + k0 + sc0);
        *(bf16x8*)&Bs[srow1 * LDSK + sc1] = *(const bf16x8*)(Bt + (long)(n0 + srow1) * ldb + k0 + sc1);
        if (k0 + 32 < K) {  // prefetch next K tile (global_prefetch_b8)
            __builtin_prefetch(A  + (long)(m0 + srow0) * lda + k0 + 32 + sc0, 0, 1);
            __builtin_prefetch(Bt + (long)(n0 + srow0) * ldb + k0 + 32 + sc1, 0, 1);
        }
        __syncthreads();

        // ---- fragments ----
        bf16x16 af[2], bfr[4];
        #pragma unroll
        for (int mt = 0; mt < 2; ++mt) {
            // A frag: lanes 0-15 hold K {hi*8.., 16+hi*8..} of row l16
            const bf16x8* rp = (const bf16x8*)&As[(wm + mt * 16 + l16) * LDSK];
            bf16x8 lo = rp[hi], hv = rp[2 + hi];
            #pragma unroll
            for (int i = 0; i < 8; ++i) { af[mt][i] = lo[i]; af[mt][i + 8] = hv[i]; }
        }
        #pragma unroll
        for (int nt = 0; nt < 4; ++nt) {
            // B frag: lane = column, K {hi*8.., 16+hi*8..} per half-wave
            const bf16x8* rp = (const bf16x8*)&Bs[(wn + nt * 16 + l16) * LDSK];
            bf16x8 lo = rp[hi], hv = rp[2 + hi];
            #pragma unroll
            for (int i = 0; i < 8; ++i) { bfr[nt][i] = lo[i]; bfr[nt][i + 8] = hv[i]; }
        }
        #pragma unroll
        for (int mt = 0; mt < 2; ++mt)
            #pragma unroll
            for (int nt = 0; nt < 4; ++nt)
                acc[mt][nt] = __builtin_amdgcn_wmma_f32_16x16x32_bf16(
                    false, af[mt], false, bfr[nt], (short)0, acc[mt][nt], false, false);
        __syncthreads();
    }

    // ---- epilogue: lane l16 = N, VGPR v -> M = hi*8 + v ----
    float* Cfb = OUTF32 ? Cf + (long)bz * sC : nullptr;
    #pragma unroll
    for (int mt = 0; mt < 2; ++mt) {
        #pragma unroll
        for (int nt = 0; nt < 4; ++nt) {
            int gn = n0 + wn + nt * 16 + l16;
            #pragma unroll
            for (int v = 0; v < 8; ++v) {
                int gm = m0 + wm + mt * 16 + hi * 8 + v;
                float val = acc[mt][nt][v] * scale;
                if (BIASRELU) val = fmaxf(val + bias[gn], 0.0f);
                if (CAUSAL && gn > gm) val = -__builtin_inff();
                if (OUTF32)  Cfb[(long)gm * ldc + gn] = val;
                if (OUTB16)  Cb[(long)gm * ldc + gn] = to_bf16(val);
                if (OUTB16T) Cbt[((long)(gm / trows) * N + gn) * trows + (gm % trows)] = to_bf16(val);
            }
        }
    }
}

// ---- elementwise f32 -> bf16 (n multiple of 2048) ----
__global__ __launch_bounds__(256)
void cvt_bf16(const float* __restrict__ x, __bf16* __restrict__ y)
{
    long i = ((long)blockIdx.x * 256 + threadIdx.x) * 8;
    f32x4 a = *(const f32x4*)(x + i);
    f32x4 b = *(const f32x4*)(x + i + 4);
    bf16x8 o;
    #pragma unroll
    for (int j = 0; j < 4; ++j) { o[j] = to_bf16(a[j]); o[4 + j] = to_bf16(b[j]); }
    *(bf16x8*)(y + i) = o;
}

// ---- transpose + convert: out[c][r] = bf16(in[r][c]), R,C multiples of 32 ----
__global__ __launch_bounds__(256)
void transpose_cvt(const float* __restrict__ in, __bf16* __restrict__ out, int R, int C)
{
    __shared__ float t[32][33];
    int bx = blockIdx.x * 32, by = blockIdx.y * 32;
    int tx = threadIdx.x & 31, ty = threadIdx.x >> 5;
    #pragma unroll
    for (int i = 0; i < 4; ++i)
        t[ty + i * 8][tx] = in[(long)(by + ty + i * 8) * C + bx + tx];
    __syncthreads();
    #pragma unroll
    for (int i = 0; i < 4; ++i)
        out[(long)(bx + ty + i * 8) * R + by + tx] = to_bf16(t[tx][ty + i * 8]);
}

// ---- softmax over 2048-wide f32 rows; writes bf16 P in place over the row ----
__global__ __launch_bounds__(256)
void softmax_rows_bf16(float* __restrict__ S)
{
    float*  p = S + (long)blockIdx.x * 2048;
    __bf16* q = (__bf16*)p;           // bf16 result overlays front half of own row
    int tid = threadIdx.x;
    __shared__ float red[256];

    f32x4 a = *(const f32x4*)(p + tid * 8);
    f32x4 b = *(const f32x4*)(p + tid * 8 + 4);
    float v[8], mx = -__builtin_inff();
    #pragma unroll
    for (int i = 0; i < 4; ++i) { v[i] = a[i]; v[4 + i] = b[i]; }
    #pragma unroll
    for (int i = 0; i < 8; ++i) mx = fmaxf(mx, v[i]);

    red[tid] = mx; __syncthreads();
    for (int s = 128; s > 0; s >>= 1) {
        if (tid < s) red[tid] = fmaxf(red[tid], red[tid + s]);
        __syncthreads();
    }
    mx = red[0]; __syncthreads();

    float sum = 0.0f;
    #pragma unroll
    for (int i = 0; i < 8; ++i) { v[i] = __expf(v[i] - mx); sum += v[i]; }
    red[tid] = sum; __syncthreads();
    for (int s = 128; s > 0; s >>= 1) {
        if (tid < s) red[tid] += red[tid + s];
        __syncthreads();
    }
    float inv = 1.0f / red[0];

    bf16x8 o;
    #pragma unroll
    for (int i = 0; i < 8; ++i) o[i] = to_bf16(v[i] * inv);
    *(bf16x8*)(q + tid * 8) = o;
}

// ---- Y = [relu](LayerNorm(X + R)); optional bf16 copy Yb; rows of 1024 ----
__global__ __launch_bounds__(256)
void add_ln(const float* __restrict__ X, const float* __restrict__ R,
            const float* __restrict__ gamma, const float* __restrict__ beta,
            float* __restrict__ Y, __bf16* __restrict__ Yb, int relu_out)
{
    long row = blockIdx.x;
    int tid = threadIdx.x;
    f32x4 v = ((const f32x4*)(X + row * 1024))[tid] + ((const f32x4*)(R + row * 1024))[tid];

    float s = 0.0f, s2 = 0.0f;
    #pragma unroll
    for (int i = 0; i < 4; ++i) { s += v[i]; s2 += v[i] * v[i]; }

    __shared__ float rs[256], rs2[256];
    rs[tid] = s; rs2[tid] = s2; __syncthreads();
    for (int k = 128; k > 0; k >>= 1) {
        if (tid < k) { rs[tid] += rs[tid + k]; rs2[tid] += rs2[tid + k]; }
        __syncthreads();
    }
    float mu  = rs[0] * (1.0f / 1024.0f);
    float var = rs2[0] * (1.0f / 1024.0f) - mu * mu;
    float inv = rsqrtf(var + 1e-3f);

    f32x4 g = ((const f32x4*)gamma)[tid];
    f32x4 b = ((const f32x4*)beta)[tid];
    f32x4 o;
    bf16x4 ob;
    #pragma unroll
    for (int i = 0; i < 4; ++i) {
        float t = g[i] * (v[i] - mu) * inv + b[i];
        if (relu_out) t = fmaxf(t, 0.0f);
        o[i] = t; ob[i] = to_bf16(t);
    }
    ((f32x4*)(Y + row * 1024))[tid] = o;
    if (Yb) ((bf16x4*)(Yb + row * 1024))[tid] = ob;
}

extern "C" void kernel_launch(void* const* d_in, const int* in_sizes, int n_in,
                              void* d_out, int out_size, void* d_ws, size_t ws_size,
                              hipStream_t stream)
{
    (void)in_sizes; (void)n_in; (void)out_size; (void)ws_size;
    const float* inputs  = (const float*)d_in[0];
    const float* context = (const float*)d_in[1];
    const float* Wf[7] = { (const float*)d_in[2], (const float*)d_in[3], (const float*)d_in[4],
                           (const float*)d_in[5], (const float*)d_in[6], (const float*)d_in[7],
                           (const float*)d_in[10] };       // Wq Wk Wv Wcq Wck Wcv W1
    const float* gamma = (const float*)d_in[8];
    const float* beta  = (const float*)d_in[9];
    const float* b1    = (const float*)d_in[11];
    float* out = (float*)d_out;

    const int Bb = 8, S = 2048, E = 1024;
    const long MS = (long)Bb * S;          // 16384 flattened rows
    const long sQK = (long)S * E, sSS = (long)S * S;
    const float scl = 1.0f / 32.0f;        // 1/sqrt(1024)

    // ---- workspace layout ----
    char* w = (char*)d_ws;
    float* Sc = (float*)w;  w += (long)Bb * S * S * 4;   // scores f32 (128MB); bf16 P overlays rows
    float* F0 = (float*)w;  w += MS * E * 4;             // gemm f32 out scratch
    float* H1 = (float*)w;  w += MS * E * 4;
    float* H2 = (float*)w;  w += MS * E * 4;
    __bf16* Xb  = (__bf16*)w; w += MS * E * 2;           // inputs bf16
    __bf16* Cxb = (__bf16*)w; w += MS * E * 2;           // context bf16
    __bf16* Qb  = (__bf16*)w; w += MS * E * 2;
    __bf16* Kb  = (__bf16*)w; w += MS * E * 2;
    __bf16* Vtb = (__bf16*)w; w += MS * E * 2;           // V^T per batch: [b][E][S]
    __bf16* H1b = (__bf16*)w; w += MS * E * 2;
    __bf16* H2b = (__bf16*)w; w += MS * E * 2;
    __bf16* Wt[7];
    for (int i = 0; i < 7; ++i) { Wt[i] = (__bf16*)w; w += (long)E * E * 2; }

    const dim3 blk(256);
    const dim3 gP(E / 128, MS / 128, 1);   // dense projections / FFN
    const dim3 gS(S / 128, S / 128, Bb);   // attention scores (batched)
    const dim3 gO(E / 128, S / 128, Bb);   // P x V (batched)
    const dim3 gT(E / 32, E / 32, 1);
    const unsigned rows = (unsigned)MS;
    __bf16* Pb = (__bf16*)Sc;              // bf16 P, row stride 4096 elements

    // ---- precompute bf16 operands ----
    cvt_bf16<<<dim3((unsigned)(MS * E / 2048)), blk, 0, stream>>>(inputs,  Xb);
    cvt_bf16<<<dim3((unsigned)(MS * E / 2048)), blk, 0, stream>>>(context, Cxb);
    for (int i = 0; i < 7; ++i)
        transpose_cvt<<<gT, blk, 0, stream>>>(Wf[i], Wt[i], E, E);

    // ---- stage 1: causal self-attention ----
    gemm_bf16_nt<0,0,0,1,0><<<gP, blk, 0, stream>>>(Xb, Wt[0], nullptr, nullptr, Qb, nullptr,
        (int)MS, E, E, E, E, E, 0, 0, 0, 1.0f, 0);
    gemm_bf16_nt<0,0,0,1,0><<<gP, blk, 0, stream>>>(Xb, Wt[1], nullptr, nullptr, Kb, nullptr,
        (int)MS, E, E, E, E, E, 0, 0, 0, 1.0f, 0);
    gemm_bf16_nt<0,0,0,0,1><<<gP, blk, 0, stream>>>(Xb, Wt[2], nullptr, nullptr, nullptr, Vtb,
        (int)MS, E, E, E, E, E, 0, 0, 0, 1.0f, S);
    gemm_bf16_nt<1,0,1,0,0><<<gS, blk, 0, stream>>>(Qb, Kb, nullptr, Sc, nullptr, nullptr,
        S, S, E, E, E, S, sQK, sQK, sSS, scl, 0);
    softmax_rows_bf16<<<dim3(rows), blk, 0, stream>>>(Sc);
    gemm_bf16_nt<0,0,1,0,0><<<gO, blk, 0, stream>>>(Pb, Vtb, nullptr, F0, nullptr, nullptr,
        S, E, S, 2 * S, S, E, (long)S * 2 * S, sQK, sQK, 1.0f, 0);
    add_ln<<<dim3(rows), blk, 0, stream>>>(F0, inputs, gamma, beta, H1, H1b, 0);

    // ---- stage 2: cross-attention ----
    gemm_bf16_nt<0,0,0,1,0><<<gP, blk, 0, stream>>>(H1b, Wt[3], nullptr, nullptr, Qb, nullptr,
        (int)MS, E, E, E, E, E, 0, 0, 0, 1.0f, 0);
    gemm_bf16_nt<0,0,0,1,0><<<gP, blk, 0, stream>>>(Cxb, Wt[4], nullptr, nullptr, Kb, nullptr,
        (int)MS, E, E, E, E, E, 0, 0, 0, 1.0f, 0);
    gemm_bf16_nt<0,0,0,0,1><<<gP, blk, 0, stream>>>(Cxb, Wt[5], nullptr, nullptr, nullptr, Vtb,
        (int)MS, E, E, E, E, E, 0, 0, 0, 1.0f, S);
    gemm_bf16_nt<0,0,1,0,0><<<gS, blk, 0, stream>>>(Qb, Kb, nullptr, Sc, nullptr, nullptr,
        S, S, E, E, E, S, sQK, sQK, sSS, scl, 0);
    softmax_rows_bf16<<<dim3(rows), blk, 0, stream>>>(Sc);
    gemm_bf16_nt<0,0,1,0,0><<<gO, blk, 0, stream>>>(Pb, Vtb, nullptr, F0, nullptr, nullptr,
        S, E, S, 2 * S, S, E, (long)S * 2 * S, sQK, sQK, 1.0f, 0);
    add_ln<<<dim3(rows), blk, 0, stream>>>(F0, H1, gamma, beta, H2, H2b, 0);

    // ---- stage 3: FFN + final LN + relu ----
    gemm_bf16_nt<0,1,1,0,0><<<gP, blk, 0, stream>>>(H2b, Wt[6], b1, F0, nullptr, nullptr,
        (int)MS, E, E, E, E, E, 0, 0, 0, 1.0f, 0);
    add_ln<<<dim3(rows), blk, 0, stream>>>(F0, H2, gamma, beta, out, nullptr, 1);
}
